// HeteroGATv2_48412871361055
// MI455X (gfx1250) — compile-verified, matrix operand order checked by reading, not run on hardware
//
#include <hip/hip_runtime.h>
#include <cstdint>
#include <cstddef>

// ---------------------------------------------------------------------------
// Types for CDNA5 WMMA (gfx1250, wave32)
// ---------------------------------------------------------------------------
typedef float         v8f   __attribute__((ext_vector_type(8)));
typedef unsigned int  v8u   __attribute__((ext_vector_type(8)));
typedef __bf16        v16bf __attribute__((ext_vector_type(16)));

#define HID 256
#define NHEAD 4
#define EMB 128
#define LN_EPS 1e-5f

// ---------------------------------------------------------------------------
// helpers
// ---------------------------------------------------------------------------
// pack two f32 -> bf16 pair (round-half-up) in 3 VALU ops via v_perm_b32:
// result = { hi16(b+0x8000), hi16(a+0x8000) }
static __device__ inline uint32_t bf16pack(float a, float b) {
  uint32_t ua = __builtin_bit_cast(uint32_t, a) + 0x8000u;
  uint32_t ub = __builtin_bit_cast(uint32_t, b) + 0x8000u;
  return __builtin_amdgcn_perm(ub, ua, 0x07060302u);  // bytes {ub3,ub2,ua3,ua2}
}

// monotone float <-> uint encoding so atomicMax(uint) == float max
static __device__ inline unsigned fenc(float f) {
  unsigned u = __builtin_bit_cast(unsigned, f);
  return (u & 0x80000000u) ? ~u : (u | 0x80000000u);
}
static __device__ inline float fdec(unsigned u) {
  unsigned v = (u & 0x80000000u) ? (u & 0x7FFFFFFFu) : ~u;
  return __builtin_bit_cast(float, v);
}

// ---------------------------------------------------------------------------
// GEMM: C[M,N] = A[M,K] (f32) * B[K,N] (f32) (+ bias[N]),  bf16 WMMA, f32 acc
// Block: 128 threads (4 waves). Block tile 64(M) x 128(N), K-step 32.
// Pipelined: LDS tiles double-buffered; global loads for tile k+1 issue
// before the 8 WMMAs of tile k and are packed into the alternate buffer
// afterwards (one barrier per K-step, s_wait_loadcnt hidden under WMMAs).
// Operands live in LDS in fragment-major pair order so every fragment is two
// ds_load_b128 into contiguous VGPRs feeding v_wmma directly:
//   frag lane l<16 : row=l,    K pairs j=0..3 and 8..11
//   frag lane l>=16: row=l-16, K pairs j=4..7 and 12..15
// Requires: K % 32 == 0, N % 128 == 0 (true here: N in {128,256,1024}).
// ---------------------------------------------------------------------------
__global__ __launch_bounds__(128) void gemm_bf16_wmma(
    const float* __restrict__ A, const float* __restrict__ B,
    const float* __restrict__ bias, float* __restrict__ C,
    int M, int N, int K)
{
  __shared__ uint32_t lA[2][64][20];    // [buf][m-row][kpair], pad 20
  __shared__ uint32_t lB[2][128][20];   // [buf][n-col][kpair]

  const int tid  = threadIdx.x;
  const int wave = tid >> 5;
  const int lane = tid & 31;
  const int l16  = lane & 15;
  const int hi16 = lane >> 4;
  const int mbase = blockIdx.x * 64;
  const int nbase = blockIdx.y * 128;
  const int arow  = wave * 16 + l16;

  // fixed per-thread staging coordinates
  const int aRow  = tid >> 1;
  const int aHalf = tid & 1;
  int gr = mbase + aRow; if (gr > M - 1) gr = M - 1;
  const float* aPtr = A + (size_t)gr * K + aHalf * 16;
  const int nc = lane * 4;          // 0..124
  const int j0 = tid >> 5;          // 0..3
  const float* bPtr = B + nbase + nc;

  float4 ra[4];
  float4 rb0[4], rb1[4];

  auto gload = [&](int k0) {
    const float4* pa_ = (const float4*)(aPtr + k0);
#pragma unroll
    for (int i = 0; i < 4; ++i) ra[i] = pa_[i];
#pragma unroll
    for (int s = 0; s < 4; ++s) {
      int j = j0 + s * 4;
      rb0[s] = *(const float4*)(bPtr + (size_t)(k0 + 2 * j)     * N);
      rb1[s] = *(const float4*)(bPtr + (size_t)(k0 + 2 * j + 1) * N);
    }
  };
  auto lstore = [&](int buf) {
    uint32_t* d = &lA[buf][aRow][aHalf * 8];
#pragma unroll
    for (int i = 0; i < 4; ++i) {
      d[2 * i + 0] = bf16pack(ra[i].x, ra[i].y);
      d[2 * i + 1] = bf16pack(ra[i].z, ra[i].w);
    }
#pragma unroll
    for (int s = 0; s < 4; ++s) {
      int j = j0 + s * 4;
      lB[buf][nc + 0][j] = bf16pack(rb0[s].x, rb1[s].x);
      lB[buf][nc + 1][j] = bf16pack(rb0[s].y, rb1[s].y);
      lB[buf][nc + 2][j] = bf16pack(rb0[s].z, rb1[s].z);
      lB[buf][nc + 3][j] = bf16pack(rb0[s].w, rb1[s].w);
    }
  };

  v8f acc[8] = {};

  gload(0);
  lstore(0);
  __syncthreads();

  int buf = 0;
  for (int k0 = 0; k0 < K; k0 += 32) {
    const bool more = (k0 + 32) < K;
    if (more) gload(k0 + 32);               // global loads in flight during WMMAs

    const uint4* pa = (const uint4*)&lA[buf][arow][0];
    uint4 alo = pa[hi16];       // lane<16: j0..3   lane>=16: j4..7
    uint4 ahi = pa[hi16 + 2];   // lane<16: j8..11  lane>=16: j12..15
    v8u af = { alo.x, alo.y, alo.z, alo.w, ahi.x, ahi.y, ahi.z, ahi.w };
    v16bf aF = __builtin_bit_cast(v16bf, af);

#pragma unroll
    for (int ns = 0; ns < 8; ++ns) {
      const uint4* pb = (const uint4*)&lB[buf][ns * 16 + l16][0];
      uint4 blo = pb[hi16];
      uint4 bhi = pb[hi16 + 2];
      v8u bu = { blo.x, blo.y, blo.z, blo.w, bhi.x, bhi.y, bhi.z, bhi.w };
      v16bf bF = __builtin_bit_cast(v16bf, bu);
      acc[ns] = __builtin_amdgcn_wmma_f32_16x16x32_bf16(
          false, aF, false, bF, (short)0, acc[ns], false, false);
    }

    if (more) lstore(buf ^ 1);               // pack into alternate buffer
    __syncthreads();
    buf ^= 1;
  }

#pragma unroll
  for (int ns = 0; ns < 8; ++ns) {
    int col = nbase + ns * 16 + l16;
    float bv = bias ? bias[col] : 0.0f;
#pragma unroll
    for (int r = 0; r < 8; ++r) {
      int row = mbase + wave * 16 + r + hi16 * 8;   // C layout: lane>=16 -> M=r+8
      if (row < M) C[(size_t)row * N + col] = acc[ns][r] + bv;
    }
  }
}

// ---------------------------------------------------------------------------
// rowwise LayerNorm (+ELU) over 256-wide rows; 1 block per row, 256 threads
// ---------------------------------------------------------------------------
static __device__ inline float block_ln_elu(float v, int t,
                                            const float* __restrict__ g,
                                            const float* __restrict__ b)
{
  __shared__ float red[8];
  float s = v;
#pragma unroll
  for (int o = 16; o > 0; o >>= 1) s += __shfl_xor(s, o, 32);
  if ((t & 31) == 0) red[t >> 5] = s;
  __syncthreads();
  float sum = 0.f;
#pragma unroll
  for (int i = 0; i < 8; ++i) sum += red[i];
  float mu = sum * (1.0f / 256.0f);
  float dv = v - mu;
  __syncthreads();
  float s2 = dv * dv;
#pragma unroll
  for (int o = 16; o > 0; o >>= 1) s2 += __shfl_xor(s2, o, 32);
  if ((t & 31) == 0) red[t >> 5] = s2;
  __syncthreads();
  float var = 0.f;
#pragma unroll
  for (int i = 0; i < 8; ++i) var += red[i];
  var *= (1.0f / 256.0f);
  float y = dv * rsqrtf(var + LN_EPS) * g[t] + b[t];
  return y > 0.f ? y : (__expf(y) - 1.f);   // ELU(alpha=1)
}

__global__ __launch_bounds__(256) void ln_elu_kernel(
    float* __restrict__ x, const float* __restrict__ g, const float* __restrict__ b)
{
  int row = blockIdx.x, t = threadIdx.x;
  float v = x[(size_t)row * HID + t];
  x[(size_t)row * HID + t] = block_ln_elu(v, t, g, b);
}

// h_new = elu(ln((acc + sum(bias_rel))/cnt + h_old))
__global__ __launch_bounds__(256) void combine_ln_kernel(
    const float* __restrict__ acc,
    const float* __restrict__ b1, const float* __restrict__ b2,
    const float* __restrict__ b3, float invcnt,
    const float* __restrict__ hold,
    const float* __restrict__ g, const float* __restrict__ bt,
    float* __restrict__ out)
{
  int row = blockIdx.x, t = threadIdx.x;
  float bs = b1[t] + b2[t] + (b3 ? b3[t] : 0.f);
  size_t idx = (size_t)row * HID + t;
  float v = (acc[idx] + bs) * invcnt + hold[idx];
  out[idx] = block_ln_elu(v, t, g, bt);
}

// ---------------------------------------------------------------------------
// edge kernels. edge id e < nE -> (src[e], dst[e]); else self-loop (e-nE).
// one wave32 per (edge, head)
// ---------------------------------------------------------------------------
__global__ __launch_bounds__(256) void edge_score_kernel(
    const float* __restrict__ xl, const float* __restrict__ xr,
    const int* __restrict__ src, const int* __restrict__ dst,
    int nE, int nSelf, const float* __restrict__ att, int C,
    float* __restrict__ score, unsigned* __restrict__ mEnc)
{
  int gw = blockIdx.x * 8 + (threadIdx.x >> 5);
  int lane = threadIdx.x & 31;
  int total = nE + nSelf;
  int e = gw >> 2, h = gw & 3;
  if (e >= total) return;
  int s, d;
  if (e < nE) { s = src[e]; d = dst[e]; } else { s = d = e - nE; }
  const int HC = NHEAD * C;
  const float* pl = xl + (size_t)s * HC + h * C;
  const float* pr = xr + (size_t)d * HC + h * C;
  const float* pa = att + h * C;
  float acc = 0.f;
  for (int c = lane; c < C; c += 32) {
    float v = pl[c] + pr[c];
    v = v > 0.f ? v : 0.2f * v;          // leaky_relu(0.2)
    acc += v * pa[c];
  }
#pragma unroll
  for (int o = 16; o > 0; o >>= 1) acc += __shfl_xor(acc, o, 32);
  if (lane == 0) {
    score[(size_t)e * NHEAD + h] = acc;
    atomicMax(&mEnc[(size_t)d * NHEAD + h], fenc(acc));
  }
}

__global__ __launch_bounds__(256) void edge_exp_kernel(
    float* __restrict__ score,        // in: score, out: exp(score - m[dst])
    const int* __restrict__ dst, int nE, int nSelf,
    const unsigned* __restrict__ mEnc, float* __restrict__ den)
{
  long long i = (long long)blockIdx.x * 256 + threadIdx.x;
  long long total = (long long)(nE + nSelf) * NHEAD;
  if (i >= total) return;
  int e = (int)(i >> 2), h = (int)(i & 3);
  int d = (e < nE) ? dst[e] : e - nE;
  float m = fdec(mEnc[(size_t)d * NHEAD + h]);
  float v = __expf(score[i] - m);
  score[i] = v;
  atomicAdd(&den[(size_t)d * NHEAD + h], v);
}

__global__ __launch_bounds__(256) void edge_scatter_kernel(
    const float* __restrict__ xl, const float* __restrict__ ex,
    const float* __restrict__ den,
    const int* __restrict__ src, const int* __restrict__ dst,
    int nE, int nSelf, int C, int concat, float* __restrict__ acc)
{
  int gw = blockIdx.x * 8 + (threadIdx.x >> 5);
  int lane = threadIdx.x & 31;
  int total = nE + nSelf;
  int e = gw >> 2, h = gw & 3;
  if (e >= total) return;
  int s, d;
  if (e < nE) { s = src[e]; d = dst[e]; } else { s = d = e - nE; }
  float alpha = ex[(size_t)e * NHEAD + h] / den[(size_t)d * NHEAD + h];
  if (!concat) alpha *= (1.0f / NHEAD);                  // head-mean
  const int HC = NHEAD * C;
  const float* pl = xl + (size_t)s * HC + h * C;
  float* pd = acc + (size_t)d * HID + (concat ? h * C : 0);
  for (int c = lane; c < C; c += 32)
    atomicAdd(&pd[c], pl[c] * alpha);
}

__global__ void fill_zero_kernel(uint32_t* __restrict__ p, long long n) {
  long long i = (long long)blockIdx.x * 256 + threadIdx.x;
  if (i < n) p[i] = 0u;
}

// ---------------------------------------------------------------------------
// host orchestration
// ---------------------------------------------------------------------------
static inline void launch_gemm(const float* A, const float* B, const float* bias,
                               float* C, int M, int N, int K, hipStream_t s) {
  dim3 grid((unsigned)((M + 63) / 64), (unsigned)(N / 128));
  gemm_bf16_wmma<<<grid, dim3(128), 0, s>>>(A, B, bias, C, M, N, K);
}
static inline void launch_fill0(void* p, long long n_u32, hipStream_t s) {
  fill_zero_kernel<<<(unsigned)((n_u32 + 255) / 256), 256, 0, s>>>((uint32_t*)p, n_u32);
}

extern "C" void kernel_launch(void* const* d_in, const int* in_sizes, int n_in,
                              void* d_out, int out_size, void* d_ws, size_t ws_size,
                              hipStream_t stream)
{
  (void)n_in; (void)out_size; (void)ws_size;
  // setup_inputs() dict order, params flattened in insertion order
  const float* x_outfit = (const float*)d_in[0];
  const float* x_item   = (const float*)d_in[1];
  const int* ews = (const int*)d_in[2];
  const int* ewd = (const int*)d_in[3];
  const int* ess = (const int*)d_in[4];
  const int* esd = (const int*)d_in[5];
  const int* eas = (const int*)d_in[6];
  const int* ead = (const int*)d_in[7];
  const int* ecs = (const int*)d_in[8];
  const int* ecd = (const int*)d_in[9];
  const float* Wo    = (const float*)d_in[10];
  const float* bo    = (const float*)d_in[11];
  const float* go    = (const float*)d_in[12];
  const float* beo   = (const float*)d_in[13];
  const float* Wi    = (const float*)d_in[14];
  const float* bi    = (const float*)d_in[15];
  const float* gi    = (const float*)d_in[16];
  const float* bei   = (const float*)d_in[17];
  const float* Wl0   = (const float*)d_in[18];
  const float* Wr0   = (const float*)d_in[19];
  const float* att0  = (const float*)d_in[20];
  const float* bias0 = (const float*)d_in[21];
  const float* Wl1   = (const float*)d_in[22];
  const float* Wr1   = (const float*)d_in[23];
  const float* att1  = (const float*)d_in[24];
  const float* bias1 = (const float*)d_in[25];
  const float* ng0   = (const float*)d_in[26];
  const float* nb0   = (const float*)d_in[27];
  const float* ng1   = (const float*)d_in[28];
  const float* nb1   = (const float*)d_in[29];
  const float* We    = (const float*)d_in[30];
  const float* be    = (const float*)d_in[31];

  const int OD = 512;
  const int N_O = in_sizes[0] / OD;
  const int N_I = in_sizes[1] / OD;
  const int E_W = in_sizes[2], E_S = in_sizes[4], E_A = in_sizes[6], E_C = in_sizes[8];
  const int NMAX = (N_O > N_I) ? N_O : N_I;
  int maxTot = E_W;
  if (E_S + N_O > maxTot) maxTot = E_S + N_O;
  if (E_A + N_O > maxTot) maxTot = E_A + N_O;
  if (E_C + N_I > maxTot) maxTot = E_C + N_I;

  // workspace carve
  char* ws = (char*)d_ws;
  size_t off = 0;
  auto carve = [&](size_t bytes) -> void* {
    void* p = ws + off; off += (bytes + 255) & ~(size_t)255; return p;
  };
  float*    ho_a  = (float*)carve((size_t)N_O * HID * 4);
  float*    ho_b  = (float*)carve((size_t)N_O * HID * 4);
  float*    hi_a  = (float*)carve((size_t)N_I * HID * 4);
  float*    hi_b  = (float*)carve((size_t)N_I * HID * 4);
  float*    acc_o = (float*)carve((size_t)N_O * HID * 4);
  float*    acc_i = (float*)carve((size_t)N_I * HID * 4);
  float*    xl    = (float*)carve((size_t)NMAX * (NHEAD * HID) * 4);
  float*    xr    = (float*)carve((size_t)NMAX * (NHEAD * HID) * 4);
  float*    score = (float*)carve((size_t)maxTot * NHEAD * 4);
  unsigned* mEnc  = (unsigned*)carve((size_t)NMAX * NHEAD * 4);
  float*    den   = (float*)carve((size_t)NMAX * NHEAD * 4);

  // ---- input projections: h = elu(ln(x @ W + b)) ----
  launch_gemm(x_outfit, Wo, bo, ho_a, N_O, HID, OD, stream);
  ln_elu_kernel<<<(unsigned)N_O, 256, 0, stream>>>(ho_a, go, beo);
  launch_gemm(x_item, Wi, bi, hi_a, N_I, HID, OD, stream);
  ln_elu_kernel<<<(unsigned)N_I, 256, 0, stream>>>(hi_a, gi, bei);

  float* ho_cur = ho_a; float* ho_nxt = ho_b;
  float* hi_cur = hi_a; float* hi_nxt = hi_b;

  for (int l = 0; l < 2; ++l) {
    const int Ch = l ? HID : (HID / NHEAD);   // 256 or 64
    const int HC = NHEAD * Ch;
    const int concat = (l == 0);
    const float* WlB   = l ? Wl1 : Wl0;
    const float* WrB   = l ? Wr1 : Wr0;
    const float* attB  = l ? att1 : att0;
    const float* biasB = l ? bias1 : bias0;
    const float* ng    = l ? ng1 : ng0;
    const float* nb    = l ? nb1 : nb0;

    launch_fill0(acc_o, (long long)N_O * HID, stream);
    launch_fill0(acc_i, (long long)N_I * HID, stream);

    struct RelDesc {
      const float* xs; const float* xd; int nsN, ndN;
      const int *s, *d; int nE, nSelf; float* acc;
    };
    const RelDesc rels[5] = {
      { ho_cur, hi_cur, N_O, N_I, ews, ewd, E_W, 0,   acc_i },  // 0: wears   (o->i)
      { hi_cur, ho_cur, N_I, N_O, ewd, ews, E_W, 0,   acc_o },  // 1: wornby  (i->o)
      { ho_cur, ho_cur, N_O, N_O, ess, esd, E_S, N_O, acc_o },  // 2: sim  (+self)
      { ho_cur, ho_cur, N_O, N_O, eas, ead, E_A, N_O, acc_o },  // 3: age  (+self)
      { hi_cur, hi_cur, N_I, N_I, ecs, ecd, E_C, N_I, acc_i },  // 4: cooc (+self)
    };

    for (int r = 0; r < 5; ++r) {
      const RelDesc& R = rels[r];
      launch_gemm(R.xs, WlB + (size_t)r * HID * HC, nullptr, xl, R.nsN, HC, HID, stream);
      launch_gemm(R.xd, WrB + (size_t)r * HID * HC, nullptr, xr, R.ndN, HC, HID, stream);
      launch_fill0(mEnc, (long long)R.ndN * NHEAD, stream);  // 0 == encoded -max
      launch_fill0(den,  (long long)R.ndN * NHEAD, stream);
      const int total = R.nE + R.nSelf;
      const long long waves = (long long)total * NHEAD;
      const unsigned wblocks = (unsigned)((waves + 7) / 8);
      edge_score_kernel<<<wblocks, 256, 0, stream>>>(
          xl, xr, R.s, R.d, R.nE, R.nSelf, attB + (size_t)r * NHEAD * Ch, Ch, score, mEnc);
      edge_exp_kernel<<<(unsigned)((waves + 255) / 256), 256, 0, stream>>>(
          score, R.d, R.nE, R.nSelf, mEnc, den);
      edge_scatter_kernel<<<wblocks, 256, 0, stream>>>(
          xl, score, den, R.s, R.d, R.nE, R.nSelf, Ch, concat, R.acc);
    }

    // ho = elu(ln((wornby+sim+age)/3 + ho)); hi = elu(ln((wears+cooc)/2 + hi))
    combine_ln_kernel<<<(unsigned)N_O, 256, 0, stream>>>(
        acc_o, biasB + 1 * HID, biasB + 2 * HID, biasB + 3 * HID, 1.0f / 3.0f,
        ho_cur, ng, nb, ho_nxt);
    combine_ln_kernel<<<(unsigned)N_I, 256, 0, stream>>>(
        acc_i, biasB + 0 * HID, biasB + 4 * HID, nullptr, 0.5f,
        hi_cur, ng, nb, hi_nxt);

    float* t;
    t = ho_cur; ho_cur = ho_nxt; ho_nxt = t;
    t = hi_cur; hi_cur = hi_nxt; hi_nxt = t;
  }

  // ---- final embedding projection straight into d_out ----
  float* outF = (float*)d_out;
  launch_gemm(ho_cur, We, be, outF, N_O, EMB, HID, stream);
  launch_gemm(hi_cur, We, be, outF + (size_t)N_O * EMB, N_I, EMB, HID, stream);
}
